// GAT_node_classification_74431783240458
// MI455X (gfx1250) — compile-verified, matrix operand order checked by reading, not run on hardware
//
#include <hip/hip_runtime.h>
#include <hip/hip_bf16.h>

// ---------------------------------------------------------------------------
// GAT node classification for MI455X (gfx1250, wave32, WMMA).
//   N=100000 nodes, F=128 in-features, H=4 heads, C=32 ch/head, HC=128, CLS=2
//   E=1.6M edges (+N self loops)
// ---------------------------------------------------------------------------

static constexpr int   NN  = 100000;
static constexpr int   NN_PAD = ((NN + 127) / 128) * 128;   // 100096 (padded h rows)
static constexpr int   FF  = 128;
static constexpr int   HH  = 4;
static constexpr int   CC  = 32;
static constexpr int   HC  = 128;   // H*C
static constexpr int   NCLS = 2;
static constexpr long long EE = 1600000;
static constexpr int   ET  = (int)EE + NN;   // edges incl. self loops = 1,700,000
static constexpr float NEG_SLOPE = 0.2f;

// workspace layout (float offsets). NOTE: h is padded to NN_PAD rows; the
// overflow rows stay inside h's own padded region (a_src starts after it),
// so no other buffer is hit — this removes all store guards from the WMMA GEMM.
static constexpr size_t OFF_H    = 0;                           // h  [NN_PAD,128]
static constexpr size_t OFF_ASRC = OFF_H    + (size_t)NN_PAD*HC;   // a_src [N,4]
static constexpr size_t OFF_ADST = OFF_ASRC + (size_t)NN*HH;       // a_dst [N,4]
static constexpr size_t OFF_M    = OFF_ADST + (size_t)NN*HH;       // seg-max [N,4]
static constexpr size_t OFF_DEN  = OFF_M    + (size_t)NN*HH;       // denom   [N,4]
static constexpr size_t OFF_ACC  = OFF_DEN  + (size_t)NN*HH;       // accum [N,128]

typedef __attribute__((ext_vector_type(16))) __bf16 v16bf;
typedef __attribute__((ext_vector_type(8)))  float  v8f;

// fp32 -> bf16 bits, round-to-nearest-even
static __device__ __forceinline__ unsigned short f2bf_bits(float f) {
    unsigned u = __builtin_bit_cast(unsigned, f);
    u += 0x7FFFu + ((u >> 16) & 1u);
    return (unsigned short)(u >> 16);
}
static __device__ __forceinline__ __bf16 bits2bf(unsigned short s) {
    return __builtin_bit_cast(__bf16, s);
}
static __device__ __forceinline__ __bf16 f2bf(float f) {
    return bits2bf(f2bf_bits(f));
}

// ---------------------------------------------------------------------------
// Kernel 1: h = x @ W_gat^T    (WMMA bf16, fp32 accumulate)
//   block = 256 threads (8 waves); block covers 128 rows; wave w -> 16 rows.
//   W staged once per block into LDS as bf16 (32 KB); B fragments are two
//   contiguous 16-byte LDS reads per k-step. A fragments cached in registers.
//
//   Fragment layout per ISA 7.12.2 (16-bit A 16x32): lane's halves 0..7 hold
//   K = kbase+0..7, halves 8..15 hold K = kbase+16..23, kbase = (lane>>4)*8.
//   B uses the symmetric layout with lane -> column, K pattern identical.
// ---------------------------------------------------------------------------
__global__ void __launch_bounds__(256) k_gemm(
        const float* __restrict__ x,
        const float* __restrict__ Wg,   // [128,128] row-major (o,f)
        float* __restrict__ h)          // [NN_PAD,128]
{
    __shared__ unsigned short wlds[HC * FF];     // W as bf16, row-major (o,f)

    // cooperative fp32->bf16 stage of W (16384 elems / 256 thr = 64 each)
    for (int i = threadIdx.x; i < HC * FF; i += 256)
        wlds[i] = f2bf_bits(Wg[i]);
    __syncthreads();

    const int lane = threadIdx.x & 31;
    const int wid  = threadIdx.x >> 5;
    const int lo   = lane & 15;
    const int hi   = lane >> 4;
    const int r0   = blockIdx.x * 128 + wid * 16;

    int arow = r0 + lo;
    if (arow >= NN) arow = NN - 1;          // clamp loads only; stores unguarded
    const float* ap = x + (size_t)arow * FF + hi * 8;

    // A fragments for the whole K=128 (two aligned float4 loads per group)
    v16bf afr[4];
#pragma unroll
    for (int ks = 0; ks < 4; ++ks) {
        const float4 q0 = *(const float4*)(ap + ks * 32 + 0);
        const float4 q1 = *(const float4*)(ap + ks * 32 + 4);
        const float4 q2 = *(const float4*)(ap + ks * 32 + 16);
        const float4 q3 = *(const float4*)(ap + ks * 32 + 20);
        afr[ks][0]  = f2bf(q0.x); afr[ks][1]  = f2bf(q0.y);
        afr[ks][2]  = f2bf(q0.z); afr[ks][3]  = f2bf(q0.w);
        afr[ks][4]  = f2bf(q1.x); afr[ks][5]  = f2bf(q1.y);
        afr[ks][6]  = f2bf(q1.z); afr[ks][7]  = f2bf(q1.w);
        afr[ks][8]  = f2bf(q2.x); afr[ks][9]  = f2bf(q2.y);
        afr[ks][10] = f2bf(q2.z); afr[ks][11] = f2bf(q2.w);
        afr[ks][12] = f2bf(q3.x); afr[ks][13] = f2bf(q3.y);
        afr[ks][14] = f2bf(q3.z); afr[ks][15] = f2bf(q3.w);
    }

    const unsigned short* bl = wlds + lo * FF + hi * 8;     // + o0*FF per tile
    float* op = h + (size_t)(r0 + hi * 8) * HC + lo;        // + j*HC + t*16

#pragma unroll
    for (int t = 0; t < 8; ++t) {
        v8f acc = {};
#pragma unroll
        for (int ks = 0; ks < 4; ++ks) {
            const unsigned short* p = bl + t * 16 * FF + ks * 32;
            v16bf bfr;
#pragma unroll
            for (int i = 0; i < 8; ++i) {
                bfr[i]     = bits2bf(p[i]);        // 16B contiguous -> b128
                bfr[8 + i] = bits2bf(p[16 + i]);   // 16B contiguous -> b128
            }
            acc = __builtin_amdgcn_wmma_f32_16x16x32_bf16(
                    false, afr[ks], false, bfr, (short)0, acc, false, false);
        }
        // C/D layout: VGPR j, lanes 0-15 -> M=j, lanes 16-31 -> M=j+8; N=lane&15
#pragma unroll
        for (int j = 0; j < 8; ++j)
            op[(size_t)j * HC + t * 16] = acc[j];
    }
}

// ---------------------------------------------------------------------------
// Kernel 2: a_src[n,h] = sum_c h[n,h,c]*att_src[h,c]  (and a_dst). Wave/node.
// ---------------------------------------------------------------------------
__global__ void k_attn(const float* __restrict__ h,
                       const float* __restrict__ att_src,  // [4,32]
                       const float* __restrict__ att_dst,  // [4,32]
                       float* __restrict__ a_src,          // [N,4]
                       float* __restrict__ a_dst)
{
    const int lane = threadIdx.x & 31;
    const int node = (blockIdx.x * blockDim.x + threadIdx.x) >> 5;
    if (node >= NN) return;
    const float* hp = h + (size_t)node * HC;
#pragma unroll
    for (int j = 0; j < HH; ++j) {
        const float v  = hp[j * CC + lane];
        float ps = v * att_src[j * CC + lane];
        float pd = v * att_dst[j * CC + lane];
#pragma unroll
        for (int off = 16; off > 0; off >>= 1) {
            ps += __shfl_xor(ps, off, 32);
            pd += __shfl_xor(pd, off, 32);
        }
        if (lane == 0) {
            a_src[node * HH + j] = ps;
            a_dst[node * HH + j] = pd;
        }
    }
}

// ---------------------------------------------------------------------------
// Kernel 3: reset segment-max / denom / accum (runs every launch).
// ---------------------------------------------------------------------------
__global__ void k_init(float* __restrict__ m, float* __restrict__ den,
                       float* __restrict__ acc)
{
    const int tid = blockIdx.x * blockDim.x + threadIdx.x;
    if (tid < NN * HH) {
        m[tid]   = -__builtin_inff();
        den[tid] = 0.0f;
    }
    if (tid < NN * HC) acc[tid] = 0.0f;
}

static __device__ __forceinline__ float leaky(float e) {
    return e >= 0.0f ? e : NEG_SLOPE * e;
}

// order-preserving float atomic max (init must be -inf)
static __device__ __forceinline__ void atomicMaxF(float* addr, float val) {
    if (val >= 0.0f) atomicMax((int*)addr, __float_as_int(val));
    else             atomicMin((unsigned int*)addr, __float_as_uint(val));
}

// ---------------------------------------------------------------------------
// Kernel 4: per-edge segment max. Thread/edge; edge id >= E -> self loop.
// ---------------------------------------------------------------------------
__global__ void k_edge_max(const long long* __restrict__ ei,  // [2,E]
                           const float* __restrict__ a_src,
                           const float* __restrict__ a_dst,
                           float* __restrict__ m)
{
    const int tid = blockIdx.x * blockDim.x + threadIdx.x;
    if (tid >= ET) return;
    int s, d;
    if (tid < (int)EE) { s = (int)ei[tid]; d = (int)ei[EE + tid]; }
    else               { s = d = tid - (int)EE; }
    const float4 as = *(const float4*)(a_src + (size_t)s * HH);
    const float4 ad = *(const float4*)(a_dst + (size_t)d * HH);
    float* mp = m + (size_t)d * HH;
    atomicMaxF(mp + 0, leaky(as.x + ad.x));
    atomicMaxF(mp + 1, leaky(as.y + ad.y));
    atomicMaxF(mp + 2, leaky(as.z + ad.z));
    atomicMaxF(mp + 3, leaky(as.w + ad.w));
}

// ---------------------------------------------------------------------------
// Kernel 5: per-edge exp + unnormalized aggregation. WAVE per edge:
//   lanes gather h[src] (coalesced 128B per head) and atomicAdd into accum[dst].
//   h (51 MB) and accum (51 MB) both fit in the 192 MB L2 -> L2-rate atomics.
// ---------------------------------------------------------------------------
__global__ void k_edge_acc(const long long* __restrict__ ei,
                           const float* __restrict__ a_src,
                           const float* __restrict__ a_dst,
                           const float* __restrict__ m,
                           const float* __restrict__ h,
                           float* __restrict__ den,
                           float* __restrict__ acc)
{
    const int lane = threadIdx.x & 31;
    const int e    = (blockIdx.x * blockDim.x + threadIdx.x) >> 5;
    if (e >= ET) return;
    int s, d;
    if (e < (int)EE) { s = (int)ei[e]; d = (int)ei[EE + e]; }
    else             { s = d = e - (int)EE; }

    const float4 as = *(const float4*)(a_src + (size_t)s * HH);  // broadcast
    const float4 ad = *(const float4*)(a_dst + (size_t)d * HH);
    const float4 mx = *(const float4*)(m     + (size_t)d * HH);
    const float ex0 = __expf(leaky(as.x + ad.x) - mx.x);
    const float ex1 = __expf(leaky(as.y + ad.y) - mx.y);
    const float ex2 = __expf(leaky(as.z + ad.z) - mx.z);
    const float ex3 = __expf(leaky(as.w + ad.w) - mx.w);

    if (lane < HH) {
        const float exl = (lane == 0) ? ex0 : (lane == 1) ? ex1
                        : (lane == 2) ? ex2 : ex3;
        atomicAdd(den + (size_t)d * HH + lane, exl);
    }

    const float* hs = h   + (size_t)s * HC;
    float*       ap = acc + (size_t)d * HC;
    atomicAdd(ap +  0 + lane, ex0 * hs[ 0 + lane]);
    atomicAdd(ap + 32 + lane, ex1 * hs[32 + lane]);
    atomicAdd(ap + 64 + lane, ex2 * hs[64 + lane]);
    atomicAdd(ap + 96 + lane, ex3 * hs[96 + lane]);
}

// ---------------------------------------------------------------------------
// Kernel 6: normalize + bias + ReLU + classifier (128 -> 2). Wave per node.
// ---------------------------------------------------------------------------
__global__ void k_final(const float* __restrict__ acc,
                        const float* __restrict__ den,
                        const float* __restrict__ b_gat,   // [128]
                        const float* __restrict__ W_lin,   // [2,128]
                        const float* __restrict__ b_lin,   // [2]
                        float* __restrict__ out)           // [N,2]
{
    const int lane = threadIdx.x & 31;
    const int node = (blockIdx.x * blockDim.x + threadIdx.x) >> 5;
    if (node >= NN) return;
    const float* ap = acc + (size_t)node * HC;
    const float* dp = den + (size_t)node * HH;
    float c0 = 0.0f, c1 = 0.0f;
#pragma unroll
    for (int j = 0; j < HH; ++j) {
        const int hc = j * CC + lane;
        float v = ap[hc] / dp[j] + b_gat[hc];
        v = v > 0.0f ? v : 0.0f;               // relu
        c0 += v * W_lin[hc];
        c1 += v * W_lin[HC + hc];
    }
#pragma unroll
    for (int off = 16; off > 0; off >>= 1) {
        c0 += __shfl_xor(c0, off, 32);
        c1 += __shfl_xor(c1, off, 32);
    }
    if (lane == 0) {
        out[(size_t)node * NCLS + 0] = c0 + b_lin[0];
        out[(size_t)node * NCLS + 1] = c1 + b_lin[1];
    }
}

// ---------------------------------------------------------------------------
extern "C" void kernel_launch(void* const* d_in, const int* in_sizes, int n_in,
                              void* d_out, int out_size, void* d_ws, size_t ws_size,
                              hipStream_t stream) {
    const float*     x      = (const float*)d_in[0];
    const long long* ei     = (const long long*)d_in[1];   // int64 [2,E]
    const float*     Wg     = (const float*)d_in[2];
    const float*     att_s  = (const float*)d_in[3];
    const float*     att_d  = (const float*)d_in[4];
    const float*     bg     = (const float*)d_in[5];
    const float*     Wl     = (const float*)d_in[6];
    const float*     bl     = (const float*)d_in[7];
    float*           out    = (float*)d_out;

    float* ws    = (float*)d_ws;
    float* h     = ws + OFF_H;
    float* a_src = ws + OFF_ASRC;
    float* a_dst = ws + OFF_ADST;
    float* m     = ws + OFF_M;
    float* den   = ws + OFF_DEN;
    float* acc   = ws + OFF_ACC;

    // 1) WMMA node transform (h padded to NN_PAD rows; overflow rows stay
    //    inside h's padded region, before a_src)
    k_gemm<<<NN_PAD / 128, 256, 0, stream>>>(x, Wg, h);
    // 2) attention logits (wave per node)
    k_attn<<<(NN * 32 + 255) / 256, 256, 0, stream>>>(h, att_s, att_d, a_src, a_dst);
    // 3) reset reduction buffers
    k_init<<<(NN * HC + 255) / 256, 256, 0, stream>>>(m, den, acc);
    // 4) segment max over edges
    k_edge_max<<<(ET + 255) / 256, 256, 0, stream>>>(ei, a_src, a_dst, m);
    // 5) exp + weighted scatter-add (wave per edge)
    k_edge_acc<<<(ET * 32 + 255) / 256, 256, 0, stream>>>(ei, a_src, a_dst, m, h, den, acc);
    // 6) normalize + relu + classifier (wave per node)
    k_final<<<(NN * 32 + 255) / 256, 256, 0, stream>>>(acc, den, bg, Wl, bl, out);
}
